// HADL_60593398612570
// MI455X (gfx1250) — matrix operand order repaired and checked by reading, not compiled, not run on hardware
//
#include <hip/hip_runtime.h>
#include <math.h>

// ---------------------------------------------------------------------------
// Problem constants (from reference): x[128,720,862], A[360,30], B[30,720],
// b[720].  out[128,720,862].
// Collapsed math:
//   G[l,p]   = (1/(360*sqrt2)) * sum_n D[n,l] * (A@B)[n,p]      (l<360)
//   out[b,p,c] = sum_l x2[b,c,l]*G[l,p] + mean[b,c]*(1-2*colsumG[p]) + bias[p]
// GEMM: M=110336, K=360(pad 384), N=720, done in bf16 WMMA with hi/lo split
// (3 products: hi*hi + hi*lo + lo*hi) -> ~fp32 accuracy, f32 accumulate.
// ---------------------------------------------------------------------------

#define BATCH  128
#define SEQL   720
#define CHN    862
#define INLEN  360
#define RANKK  30
#define PREDL  720
#define ROWS   (BATCH*CHN)   // 110336 = 862 * 128
#define NKC    12            // K chunks of 32 (K padded to 384)
#define NT_TOT 45            // total N tiles of 16 (720/16)
#define NT_WG  15            // N tiles per workgroup (N=240, grid.y=3)

typedef unsigned int   u32;
typedef unsigned short u16;
typedef __attribute__((ext_vector_type(4)))  u32    u32x4;
typedef __attribute__((ext_vector_type(8)))  float  v8f;
typedef __attribute__((ext_vector_type(16))) __bf16 v16bf;

union BF16Frag { v16bf v; u32x4 q[2]; };

__device__ __forceinline__ u16 f2bf(float f) {            // RNE float->bf16
    u32 u = __float_as_uint(f);
    u += 0x7FFFu + ((u >> 16) & 1u);
    return (u16)(u >> 16);
}
__device__ __forceinline__ float bf2f(u16 h) {
    return __uint_as_float(((u32)h) << 16);
}

// ---------------- setup kernel 1: AB = A @ B  [360,720] --------------------
__global__ void hadl_ab_kernel(const float* __restrict__ A,
                               const float* __restrict__ B,
                               float* __restrict__ AB) {
    int id = blockIdx.x * blockDim.x + threadIdx.x;
    if (id >= INLEN * PREDL) return;
    int n = id / PREDL, p = id - n * PREDL;
    float s = 0.f;
#pragma unroll
    for (int r = 0; r < RANKK; ++r) s += A[n * RANKK + r] * B[r * PREDL + p];
    AB[id] = s;
}

// ---------------- setup kernel 2: G = (D^T/(360*sqrt2)) @ AB  [360,720] ----
// D[n,l] = sqrt(2/360)*cos(pi*(l+.5)*n/360), row n=0 scaled by 1/sqrt2.
// Exact angle reduction: (2l+1)*n mod 1440, cos table in LDS.
__global__ void hadl_dctg_kernel(const float* __restrict__ AB,
                                 float* __restrict__ G) {
    __shared__ float ctab[1440];
    for (int i = threadIdx.x; i < 1440; i += blockDim.x)
        ctab[i] = cosf(3.14159265358979323846f * (float)i / 720.0f);
    __syncthreads();

    int id = blockIdx.x * blockDim.x + threadIdx.x;
    if (id < INLEN * PREDL) {
        int l = id / PREDL, p = id - l * PREDL;
        int step = 2 * l + 1;
        float s = AB[p] * 0.70710678118654752f;   // n=0 term, cos=1, /sqrt2
        int tix = 0;
        for (int n = 1; n < INLEN; ++n) {
            tix += step;
            if (tix >= 1440) tix -= 1440;
            s += ctab[tix] * AB[n * PREDL + p];
        }
        const float scale = 1.0f / (360.0f * sqrtf(360.0f)); // sqrt(2/360)/(360*sqrt2)
        G[id] = s * scale;
    }
}

// ------- setup kernel 3: pre-swizzle G into WMMA B-fragment layout ---------
// B-matrix (bf16, 32x16) lane layout: lane l, elem e: K = (e&7) + 8*(l>>4) + 16*(e>>3),
// N = l&15 (CDNA5 ISA 7.12.2 / sparse-B pattern).  K>=360 padded with zero.
__global__ void hadl_pack_kernel(const float* __restrict__ G,
                                 u16* __restrict__ PBh,
                                 u16* __restrict__ PBl) {
    int id = blockIdx.x * blockDim.x + threadIdx.x;
    if (id >= NKC * NT_TOT * 32 * 16) return;
    int e    = id & 15;
    int lane = (id >> 4) & 31;
    int rest = id >> 9;
    int u    = rest % NT_TOT;
    int tc   = rest / NT_TOT;
    int K = tc * 32 + (e & 7) + 8 * (lane >> 4) + 16 * (e >> 3);
    int N = u * 16 + (lane & 15);
    float v = (K < INLEN) ? G[K * PREDL + N] : 0.f;
    u16 h = f2bf(v);
    PBh[id] = h;
    PBl[id] = f2bf(v - bf2f(h));
}

// ------- setup kernel 4: mean-correction coefficient per output p ----------
__global__ void hadl_cmp_kernel(const float* __restrict__ G,
                                float* __restrict__ cmp) {
    int p = blockIdx.x * blockDim.x + threadIdx.x;
    if (p >= PREDL) return;
    float s = 0.f;
    for (int l = 0; l < INLEN; ++l) s += G[l * PREDL + p];
    cmp[p] = 1.0f - 2.0f * s;
}

// ---------------- main kernel: fused Haar + GEMM + epilogue ----------------
// grid = (862, 3), block = 256 (8 waves).  WG tile: 128 rows x 240 preds.
// Wave w: rows m0+w*16..+15, all 15 N-tiles (acc = 15 x v8f = 120 VGPRs).
__global__ __launch_bounds__(256) void hadl_gemm_kernel(
        const float* __restrict__ x,
        const u16* __restrict__ PBh, const u16* __restrict__ PBl,
        const float* __restrict__ cmp, const float* __restrict__ bias,
        float* __restrict__ out) {
    __shared__ u32   sXhi[128 * 16];      // 128 rows x 32 K, bf16-hi (u32 pairs)
    __shared__ u32   sXlo[128 * 16];      // bf16-lo
    __shared__ float spart[2 * 128];
    __shared__ float smean[128];
    __shared__ float sOut[80 * 128];      // epilogue transpose tile
    __shared__ float scmp[240];
    __shared__ float sbias[240];

    const int t    = threadIdx.x;
    const int bx   = blockIdx.x, by = blockIdx.y;
    const int r    = t & 127;             // row within tile (loader/store role)
    const int hf   = t >> 7;              // 0/1 half
    const int wave = t >> 5;
    const int lane = t & 31;
    const int nl   = lane & 15;
    const int g    = lane >> 4;

    if (t < 240) { scmp[t] = cmp[by * 240 + t]; sbias[t] = bias[by * 240 + t]; }

    // row -> (batch, channel); in/out share base since both are [*,720,862]
    const int grow = bx * 128 + r;
    const int bidx = grow / CHN;
    const int cidx = grow - bidx * CHN;
    const size_t base = (size_t)bidx * (SEQL * CHN) + (size_t)cidx;

    v8f acc[NT_WG];
    const v8f vzero = {0.f, 0.f, 0.f, 0.f, 0.f, 0.f, 0.f, 0.f};
#pragma unroll
    for (int i = 0; i < NT_WG; ++i) acc[i] = vzero;

    float rowsum = 0.f;
    // Per-lane base pointers for the pre-swizzled B fragments.  Within a
    // K-chunk, successive N-tiles are a constant 1024 B apart -> immediate
    // offsets, no per-tile VALU address math.
    const u32* pHlane = (const u32*)PBh + ((size_t)(by * NT_WG * 32) + lane) * 8;
    const u32* pLlane = (const u32*)PBl + ((size_t)(by * NT_WG * 32) + lane) * 8;
    // A-fragment LDS base: lane m = lane&15; K = (e&7)+16*(e>>3)+8*(lane>>4)
    const int abase = (wave * 16 + nl) * 16 + 4 * g;   // u32 units

    for (int tc = 0; tc < NKC; ++tc) {
        __syncthreads();
        // ---- stage: load raw x, Haar pair-sum, bf16 hi/lo split into LDS ---
        {
            const int lb = tc * 64 + hf * 32;
            u32* dh = &sXhi[r * 16 + hf * 8];
            u32* dl = &sXlo[r * 16 + hf * 8];
#pragma unroll
            for (int kk = 0; kk < 8; ++kk) {
                const int l0 = lb + 4 * kk;
                float v0 = (l0 + 0 < SEQL) ? x[base + (size_t)(l0 + 0) * CHN] : 0.f;
                float v1 = (l0 + 1 < SEQL) ? x[base + (size_t)(l0 + 1) * CHN] : 0.f;
                float v2 = (l0 + 2 < SEQL) ? x[base + (size_t)(l0 + 2) * CHN] : 0.f;
                float v3 = (l0 + 3 < SEQL) ? x[base + (size_t)(l0 + 3) * CHN] : 0.f;
                float a2 = v0 + v1, b2 = v2 + v3;
                rowsum += a2 + b2;
                u16 ha = f2bf(a2), hb = f2bf(b2);
                u16 la = f2bf(a2 - bf2f(ha)), lb2 = f2bf(b2 - bf2f(hb));
                dh[kk] = (u32)ha | ((u32)hb << 16);
                dl[kk] = (u32)la | ((u32)lb2 << 16);
            }
        }
        __syncthreads();
        // ---- compute: per-wave A fragments, stream B fragments, 3x WMMA ----
        BF16Frag ahi, alo;
        ahi.q[0] = *(const u32x4*)&sXhi[abase];
        ahi.q[1] = *(const u32x4*)&sXhi[abase + 8];
        alo.q[0] = *(const u32x4*)&sXlo[abase];
        alo.q[1] = *(const u32x4*)&sXlo[abase + 8];

        // K-chunk base (hoisted once; tiles use immediate offsets ut*1024 B)
        const u32* pHt = pHlane + (size_t)tc * (NT_TOT * 32 * 8);
        const u32* pLt = pLlane + (size_t)tc * (NT_TOT * 32 * 8);
        if (tc + 1 < NKC)   // warm next K-chunk's B block into near caches
            __builtin_prefetch((const void*)(pHt + NT_TOT * 32 * 8), 0, 3);
#pragma unroll
        for (int ut = 0; ut < NT_WG; ++ut) {
            BF16Frag bhi, blo;
            bhi.q[0] = *(const u32x4*)(pHt + ut * 256);
            bhi.q[1] = *(const u32x4*)(pHt + ut * 256 + 4);
            blo.q[0] = *(const u32x4*)(pLt + ut * 256);
            blo.q[1] = *(const u32x4*)(pLt + ut * 256 + 4);
            acc[ut] = __builtin_amdgcn_wmma_f32_16x16x32_bf16(
                          false, ahi.v, false, bhi.v, (short)0, acc[ut], false, false);
            acc[ut] = __builtin_amdgcn_wmma_f32_16x16x32_bf16(
                          false, ahi.v, false, blo.v, (short)0, acc[ut], false, false);
            acc[ut] = __builtin_amdgcn_wmma_f32_16x16x32_bf16(
                          false, alo.v, false, bhi.v, (short)0, acc[ut], false, false);
        }
    }

    // ---- mean per row -----------------------------------------------------
    spart[hf * 128 + r] = rowsum;
    __syncthreads();
    if (t < 128) smean[t] = (spart[t] + spart[128 + t]) * (1.0f / 720.0f);

    // ---- epilogue: 3 chunks of 80 preds, transpose via LDS, coalesced store
#pragma unroll
    for (int ch = 0; ch < 3; ++ch) {
        __syncthreads();
#pragma unroll
        for (int ut = 0; ut < 5; ++ut) {
#pragma unroll
            for (int v = 0; v < 8; ++v) {
                // C/D layout: acc[v] = row (m = v + 8g), col = nl
                sOut[(ut * 16 + nl) * 128 + wave * 16 + v + 8 * g] =
                    acc[ch * 5 + ut][v];
            }
        }
        __syncthreads();
        const float mean = smean[r];
#pragma unroll 4
        for (int it = 0; it < 40; ++it) {
            const int pl = it * 2 + hf;          // 0..79 within chunk
            const int pc = ch * 80 + pl;         // 0..239 within WG
            float val = sOut[pl * 128 + r] + mean * scmp[pc] + sbias[pc];
            out[base + (size_t)(by * 240 + pc) * CHN] = val;
        }
    }
}

// ---------------------------------------------------------------------------
extern "C" void kernel_launch(void* const* d_in, const int* in_sizes, int n_in,
                              void* d_out, int out_size, void* d_ws, size_t ws_size,
                              hipStream_t stream) {
    (void)in_sizes; (void)n_in; (void)out_size; (void)ws_size;
    const float* x    = (const float*)d_in[0];
    const float* A    = (const float*)d_in[1];
    const float* B    = (const float*)d_in[2];
    const float* bias = (const float*)d_in[3];
    float* out = (float*)d_out;

    char* ws = (char*)d_ws;                 // ~3.04 MB total, all 16B aligned
    float* AB  = (float*)(ws + 0);          // 360*720 f32   = 1,036,800 B
    float* G   = (float*)(ws + 1036800);    // 360*720 f32   = 1,036,800 B
    u16*   PBh = (u16*)  (ws + 2073600);    // 12*45*512 bf16 =  552,960 B
    u16*   PBl = (u16*)  (ws + 2626560);    //                    552,960 B
    float* cmp = (float*)(ws + 3179520);    // 720 f32

    hadl_ab_kernel  <<<(INLEN * PREDL + 255) / 256, 256, 0, stream>>>(A, B, AB);
    hadl_dctg_kernel<<<(INLEN * PREDL + 255) / 256, 256, 0, stream>>>(AB, G);
    hadl_pack_kernel<<<(NKC * NT_TOT * 512 + 255) / 256, 256, 0, stream>>>(G, PBh, PBl);
    hadl_cmp_kernel <<<(PREDL + 255) / 256, 256, 0, stream>>>(G, cmp);

    dim3 grid(ROWS / 128, 3);               // (862, 3)
    hadl_gemm_kernel<<<grid, 256, 0, stream>>>(x, PBh, PBl, cmp, bias, out);
}
